// UltraNet_ismart_3238405341689
// MI455X (gfx1250) — 8-bit matrix operand order repaired and checked by reading, compiled, not run on hardware
//
#include <hip/hip_runtime.h>
#include <math.h>

typedef __attribute__((ext_vector_type(8))) int v8i;

static __device__ __forceinline__ float sigmoidf_(float v){ return 1.f/(1.f+expf(-v)); }


__global__ void k_maxabs_tanh(const float* __restrict__ w, int n, float* __restrict__ out){
  __shared__ float red[256];
  float m = 0.f;
  for (int i = threadIdx.x; i < n; i += 256) m = fmaxf(m, fabsf(tanhf(w[i])));
  red[threadIdx.x] = m;
  __syncthreads();
  for (int s = 128; s > 0; s >>= 1){
    if ((int)threadIdx.x < s) red[threadIdx.x] = fmaxf(red[threadIdx.x], red[threadIdx.x+s]);
    __syncthreads();
  }
  if (threadIdx.x == 0) out[0] = red[0];
}

__global__ void k_dequant_w(const float* __restrict__ w, const float* __restrict__ mx,
                            float* __restrict__ out, int n, float levels){
  int i = blockIdx.x*blockDim.x + threadIdx.x;
  if (i >= n) return;
  float t = tanhf(w[i]);
  float wn = t/(2.f*mx[0]) + 0.5f;
  out[i] = (2.f*rintf(wn*levels) - levels)/levels;
}

__global__ void k_pack_w4(const float* __restrict__ w, const float* __restrict__ mx,
                          signed char* __restrict__ packed, int Cout, int Cin){
  int K = Cin*9;
  int nkt = (K+63)>>6;
  int total = Cout*nkt*64;
  int idx = blockIdx.x*blockDim.x + threadIdx.x;
  if (idx >= total) return;
  int co = idx/(nkt*64);
  int k  = idx%(nkt*64);
  int q = 0;
  if (k < K){
    int tap = k/Cin, ci = k%Cin;
    int dy = tap/3, dx = tap%3;
    float t = tanhf(w[((co*Cin+ci)*3+dy)*3+dx]);
    float wn = t/(2.f*mx[0]) + 0.5f;
    q = 2*(int)rintf(wn*15.f) - 15;
  }
  int mt=co>>4, m=co&15, kt=k>>6, kk=k&63;
  int v2=kk>>5; int r=kk&31;
  int hi=r>>4;  r&=15;
  int v1=r>>3;  r&=7;
  int v0=r>>2;  int j=r&3;
  int v=4*v2+2*v1+v0;
  packed[(size_t)(mt*nkt+kt)*1024 + (size_t)(m+16*hi)*32 + v*4 + j] = (signed char)q;
}

__global__ void k_affine(const float* __restrict__ bn, int C, float accscale,
                         float* __restrict__ st){
  int c = threadIdx.x;
  if (c >= C) return;
  float g=bn[c], b=bn[C+c], m=bn[2*C+c], v=bn[3*C+c];
  float inv = g*rsqrtf(v+1e-5f);
  st[c]   = inv*accscale;
  st[C+c] = b - m*inv;
}


template<int H, int W>
__global__ void k_conv1(const float* __restrict__ x, const float* __restrict__ wq,
                        const float* __restrict__ st, unsigned char* __restrict__ out,
                        int N){
  __shared__ float sw[432 + 32];
  for (int i = threadIdx.x; i < 432; i += 256) sw[i] = wq[i];
  if (threadIdx.x < 32) sw[432 + threadIdx.x] = st[threadIdx.x];
  __syncthreads();
  long P = (long)blockIdx.x*256 + threadIdx.x;
  long NP = (long)N*H*W;
  if (P >= NP) return;
  int xw = (int)(P % W); long t = P / W; int y = (int)(t % H); int n = (int)(t / H);
  float acc[16];
  #pragma unroll
  for (int cc = 0; cc < 16; ++cc) acc[cc] = 0.f;
  constexpr long chs = (long)H*W;
  const float* xb = x + (long)n*3*chs;
  for (int dy = 0; dy < 3; ++dy){
    int iy = y + dy - 1; if (iy < 0 || iy >= H) continue;
    for (int dx = 0; dx < 3; ++dx){
      int ix = xw + dx - 1; if (ix < 0 || ix >= W) continue;
      float v0 = xb[(long)iy*W + ix];
      float v1 = xb[chs + (long)iy*W + ix];
      float v2 = xb[2*chs + (long)iy*W + ix];
      int kb = dy*3 + dx;
      #pragma unroll
      for (int cc = 0; cc < 16; ++cc){
        acc[cc] += sw[cc*27 + kb]*v0 + sw[cc*27 + 9 + kb]*v1 + sw[cc*27 + 18 + kb]*v2;
      }
    }
  }
  unsigned char* o = out + P*16;
  #pragma unroll
  for (int cc = 0; cc < 16; ++cc){
    float yv = acc[cc]*sw[432 + cc] + sw[432 + 16 + cc];
    yv = fminf(fmaxf(yv, 0.f), 1.f);
    o[cc] = (unsigned char)(int)rintf(yv*15.f);
  }
}


template<int H, int W, int C>
__global__ void k_pool(const unsigned char* __restrict__ in, unsigned char* __restrict__ out,
                       int N){
  long idx = (long)blockIdx.x*256 + threadIdx.x;
  constexpr int Ho = H/2, Wo = W/2;
  long total = (long)N*Ho*Wo*C;
  if (idx >= total) return;
  int ch = (int)(idx % C); long t = idx / C;
  int xo = (int)(t % Wo); t /= Wo;
  int yo = (int)(t % Ho); int n = (int)(t / Ho);
  const unsigned char* b = in + (((long)n*H + (long)yo*2)*W + (long)xo*2)*C + ch;
  unsigned char a0 = b[0], a1 = b[C];
  unsigned char a2 = b[(long)W*C], a3 = b[(long)W*C + C];
  unsigned char m0 = a0 > a1 ? a0 : a1;
  unsigned char m1 = a2 > a3 ? a2 : a3;
  out[idx] = m0 > m1 ? m0 : m1;
}

template<int Cin, int Cout, int H, int W, int NT>
__global__ void __launch_bounds__(32*(Cout/16)) k_convq(
    const unsigned char* __restrict__ act, const signed char* __restrict__ wpack,
    const float* __restrict__ st, unsigned char* __restrict__ out)
{
  constexpr int K   = Cin*9;
  constexpr int NKT = (K+63)/64;
  constexpr int NW  = Cout/16;
  constexpr int NCHUNK = 64*NT;
  constexpr int CPT = NCHUNK/(32*NW);
  const int lane = threadIdx.x & 31;
  const int wv   = threadIdx.x >> 5;
  const int hi   = lane >> 4;
  const int ln16 = lane & 15;
  __shared__ unsigned char bl[2][NT*1024];

  const long pbase = (long)blockIdx.x*(16*NT);

  int l_ch[CPT], l_ln[CPT], l_ly[CPT], l_lx[CPT], l_kb[CPT];
  #pragma unroll
  for (int u = 0; u < CPT; ++u){
    int ch = threadIdx.x + u*(32*NW);
    l_ch[u] = ch;
    int t = ch >> 6, c = ch & 63;
    int d = c >> 1, s = c & 1;
    long pl = pbase + t*16 + (d & 15);
    l_lx[u] = (int)(pl % W);
    long pt = pl / W;
    l_ly[u] = (int)(pt % H);
    l_ln[u] = (int)(pt / H);
    l_kb[u] = (d>>4)*16 + s*32;
  }

  auto gather = [&](int kt, int par){
    #pragma unroll
    for (int u = 0; u < CPT; ++u){
      int klo = kt*64 + l_kb[u];
      int4 val = {0,0,0,0};
      if (klo < K){
        int tap = klo / Cin;
        int ci  = klo % Cin;
        int iy = l_ly[u] + tap/3 - 1;
        int ix = l_lx[u] + tap%3 - 1;
        if (iy >= 0 && iy < H && ix >= 0 && ix < W)
          val = *(const int4*)(act + (((long)l_ln[u]*H + iy)*W + ix)*Cin + ci);
      }
      *(int4*)(&bl[par][l_ch[u]*16]) = val;
    }
  };

  v8i acc[NT];
  #pragma unroll
  for (int t = 0; t < NT; ++t) acc[t] = (v8i){0,0,0,0,0,0,0,0};

  gather(0, 0);
  #pragma unroll
  for (int kt = 0; kt < NKT; ++kt){
    __syncthreads();
    v8i a = *(const v8i*)(wpack + (size_t)(wv*NKT + kt)*1024 + lane*32);
    if (kt + 1 < NKT){
      __builtin_prefetch(wpack + (size_t)(wv*NKT + kt + 1)*1024 + lane*32, 0, 1);
      gather(kt + 1, (kt + 1) & 1);
    }
    #pragma unroll
    for (int t = 0; t < NT; ++t){
      v8i b = *(const v8i*)(&bl[kt & 1][t*1024 + lane*32]);
      acc[t] = __builtin_amdgcn_wmma_i32_16x16x64_iu8(true, a, false, b, acc[t],
                                                      false, false);
    }
  }

  int co0 = wv*16 + hi*8;
  #pragma unroll
  for (int t = 0; t < NT; ++t){
    unsigned int lo = 0, hw = 0;
    #pragma unroll
    for (int r = 0; r < 8; ++r){
      int co = co0 + r;
      float y = (float)acc[t][r]*st[co] + st[Cout+co];
      y = fminf(fmaxf(y, 0.f), 1.f);
      unsigned int q = (unsigned int)(int)rintf(y*15.f);
      if (r < 4) lo |= q << (8*r); else hw |= q << (8*(r-4));
    }
    uint2 u; u.x = lo; u.y = hw;
    long p = pbase + t*16 + ln16;
    *(uint2*)(out + p*Cout + co0) = u;
  }
}


__global__ void k_head(const unsigned char* __restrict__ act, const float* __restrict__ w,
                       const float* __restrict__ bias, float* __restrict__ p, int N){
  int idx = blockIdx.x*256 + threadIdx.x;
  int total = N*200*36;
  if (idx >= total) return;
  int co = idx % 36; int pix = (idx/36) % 200; int n = idx/(36*200);
  const unsigned char* a = act + ((long)n*200 + pix)*64;
  float s = 0.f;
  #pragma unroll
  for (int ci = 0; ci < 64; ++ci) s += w[co*64 + ci]*(float)a[ci];
  p[((long)n*36 + co)*200 + pix] = s*(1.f/15.f) + bias[co];
}

__global__ void k_decode(const float* __restrict__ p, float* __restrict__ io,
                         float* __restrict__ p5, int N){
  int idx = blockIdx.x*256 + threadIdx.x;
  int total = N*1200;
  if (idx >= total) return;
  int xx = idx % 20; int yy = (idx/20) % 10; int a = (idx/200) % 6; int n = idx/1200;
  float v[6];
  #pragma unroll
  for (int j = 0; j < 6; ++j) v[j] = p[(((long)n*36 + a*6 + j)*10 + yy)*20 + xx];
  long o = (long)idx*6;
  #pragma unroll
  for (int j = 0; j < 6; ++j) p5[o+j] = v[j];
  io[o+0] = (sigmoidf_(v[0]) + (float)xx)*16.f;
  io[o+1] = (sigmoidf_(v[1]) + (float)yy)*16.f;
  io[o+2] = expf(v[2])*20.f;
  io[o+3] = expf(v[3])*20.f;
  io[o+4] = sigmoidf_(v[4]);
  io[o+5] = sigmoidf_(v[5]);
}


extern "C" void kernel_launch(void* const* d_in, const int* in_sizes, int n_in,
                              void* d_out, int out_size, void* d_ws, size_t ws_size,
                              hipStream_t stream){
  (void)in_sizes; (void)n_in; (void)out_size; (void)ws_size;
  const float* x = (const float*)d_in[0];
  const float* w[9];  for (int i=0;i<9;++i) w[i]  = (const float*)d_in[1+i];
  const float* b9 = (const float*)d_in[10];
  const float* bn[8]; for (int i=0;i<8;++i) bn[i] = (const float*)d_in[11+i];

  char* ws = (char*)d_ws;
  size_t cur = 0;
  auto alloc = [&](size_t bytes)->size_t{
    size_t o = cur; cur += (bytes + 255) & ~(size_t)255; return o;
  };
  size_t o_scale = alloc(9*sizeof(float));
  size_t o_st[8]; for (int i=0;i<8;++i) o_st[i] = alloc(128*sizeof(float));
  size_t o_w1q = alloc(432*sizeof(float));
  size_t o_w9q = alloc(2304*sizeof(float));
  const int Cins[7]  = {16,32,64,64,64,64,64};
  const int Couts[7] = {32,64,64,64,64,64,64};
  size_t o_pk[7];
  for (int i=0;i<7;++i){
    int K = Cins[i]*9; int nkt = (K+63)/64;
    o_pk[i] = alloc((size_t)Couts[i]*nkt*64);
  }
  size_t o_R0 = alloc(52428800);
  size_t o_R1 = alloc(13107200);
  size_t o_R2 = alloc(6553600);

  float* scales = (float*)(ws + o_scale);
  int wcnt[9] = {432, 32*16*9, 64*32*9, 64*64*9, 64*64*9, 64*64*9, 64*64*9, 64*64*9, 36*64};
  for (int i=0;i<9;++i)
    k_maxabs_tanh<<<1,256,0,stream>>>(w[i], wcnt[i], scales+i);
  k_dequant_w<<<2,256,0,stream>>>(w[0], scales+0, (float*)(ws+o_w1q), 432, 255.f);
  k_dequant_w<<<9,256,0,stream>>>(w[8], scales+8, (float*)(ws+o_w9q), 2304, 255.f);
  for (int i=0;i<7;++i){
    int K = Cins[i]*9, nkt = (K+63)/64, total = Couts[i]*nkt*64;
    k_pack_w4<<<(total+255)/256,256,0,stream>>>(w[1+i], scales+1+i,
        (signed char*)(ws+o_pk[i]), Couts[i], Cins[i]);
  }
  int bc[8] = {16,32,64,64,64,64,64,64};
  for (int i=0;i<8;++i)
    k_affine<<<1,64,0,stream>>>(bn[i], bc[i], (i==0)?1.f:(1.f/225.f), (float*)(ws+o_st[i]));

  unsigned char* R0 = (unsigned char*)(ws+o_R0);
  unsigned char* R1 = (unsigned char*)(ws+o_R1);
  unsigned char* R2 = (unsigned char*)(ws+o_R2);

  k_conv1<160,320><<<12800,256,0,stream>>>(x, (const float*)(ws+o_w1q),
                                           (const float*)(ws+o_st[0]), R0, 64);
  k_pool<160,320,16><<<51200,256,0,stream>>>(R0, R1, 64);
  k_convq<16,32,80,160,2><<<25600,64,0,stream>>>(R1, (const signed char*)(ws+o_pk[0]),
                                                 (const float*)(ws+o_st[1]), R0);
  k_pool<80,160,32><<<25600,256,0,stream>>>(R0, R2, 64);
  k_convq<32,64,40,80,2><<<6400,128,0,stream>>>(R2, (const signed char*)(ws+o_pk[1]),
                                                (const float*)(ws+o_st[2]), R0);
  k_pool<40,80,64><<<12800,256,0,stream>>>(R0, R1, 64);
  k_convq<64,64,20,40,2><<<1600,128,0,stream>>>(R1, (const signed char*)(ws+o_pk[2]),
                                                (const float*)(ws+o_st[3]), R0);
  k_pool<20,40,64><<<3200,256,0,stream>>>(R0, R2, 64);
  k_convq<64,64,10,20,2><<<400,128,0,stream>>>(R2, (const signed char*)(ws+o_pk[3]),
                                               (const float*)(ws+o_st[4]), R1);
  k_convq<64,64,10,20,2><<<400,128,0,stream>>>(R1, (const signed char*)(ws+o_pk[4]),
                                               (const float*)(ws+o_st[5]), R2);
  k_convq<64,64,10,20,2><<<400,128,0,stream>>>(R2, (const signed char*)(ws+o_pk[5]),
                                               (const float*)(ws+o_st[6]), R1);
  k_convq<64,64,10,20,2><<<400,128,0,stream>>>(R1, (const signed char*)(ws+o_pk[6]),
                                               (const float*)(ws+o_st[7]), R2);
  float* P = (float*)(ws+o_R0);
  k_head<<<(64*200*36+255)/256,256,0,stream>>>(R2, (const float*)(ws+o_w9q), b9, P, 64);
  float* io = (float*)d_out;
  float* p5 = io + (long)64*1200*6;
  k_decode<<<(64*1200+255)/256,256,0,stream>>>(P, io, p5, 64);
}
